// Model_68384469287330
// MI455X (gfx1250) — matrix operand order repaired and checked by reading, not compiled, not run on hardware
//
#include <hip/hip_runtime.h>
#include <hip/hip_bf16.h>
#include <math.h>

typedef __attribute__((ext_vector_type(16))) _Float16     v16h;
typedef __attribute__((ext_vector_type(8)))  float        v8f;
typedef __attribute__((ext_vector_type(4)))  unsigned int u32x4;
typedef __attribute__((ext_vector_type(8)))  unsigned int u32x8;

#define Bn   64
#define Ln   4096
#define Nn   128
#define Fn   2049
#define En   8
#define ROWS (Bn * Nn)   // 8192
#define KP   2080        // K padded to multiple of 32 (65 WMMA steps)
#define NT   129         // N tiles of 16
#define NP   (NT * 16)   // 2064

// ---------------------------------------------------------------------------
// 1) Transpose x[B,L,N] -> xN[B,N,L]  (coalesced both sides via LDS tile)
// ---------------------------------------------------------------------------
__global__ __launch_bounds__(256) void transpose_xn(const float* __restrict__ x,
                                                    float* __restrict__ xN) {
    __shared__ float tile[32][33];
    const int b  = blockIdx.z;
    const int l0 = blockIdx.x * 32;
    const int n0 = blockIdx.y * 32;
    const int tx = threadIdx.x, ty = threadIdx.y;   // 32 x 8
    #pragma unroll
    for (int i = 0; i < 32; i += 8) {
        int l = l0 + ty + i, n = n0 + tx;
        tile[ty + i][tx] = x[((size_t)b * Ln + l) * Nn + n];  // tile[l-l0][n-n0]
    }
    __syncthreads();
    #pragma unroll
    for (int i = 0; i < 32; i += 8) {
        int n = n0 + ty + i, l = l0 + tx;
        xN[((size_t)b * Nn + n) * Ln + l] = tile[tx][ty + i];
    }
}

// ---------------------------------------------------------------------------
// 2) Forward FFT per row, in LDS. Row is DMA'd HBM->LDS by the Tensor Data
//    Mover (D# per CDNA5 ISA ch.8: 1-row tile, 4096 x 4B), tracked by
//    TENSORcnt. Threads then fold mean/var and bit-reverse scatter from LDS.
// ---------------------------------------------------------------------------
__global__ __launch_bounds__(512) void fft_fwd(const float* __restrict__ xN,
                                               float* __restrict__ fre,
                                               float* __restrict__ fim,
                                               float* __restrict__ meanvar) {
    __shared__ float re[Ln], im[Ln];
    __shared__ float r1[512], r2[512];
    const int row = blockIdx.x;
    const int tid = threadIdx.x;
    const float* src = xN + (size_t)row * Ln;

    // --- TDM: im[0..4095] = src[0..4095] (wave 0 issues; wave-level op) ---
    if (tid < 32) {
        unsigned long long ga = (unsigned long long)(uintptr_t)src; // 57-bit global addr
        unsigned ldsOff = (unsigned)(uintptr_t)&im[0];              // LDS aperture: addr[31:0] = LDS byte offset
        u32x4 g0;
        g0[0] = 1u;                                   // count=1, is_restore=0, no gather
        g0[1] = ldsOff;                               // lds_addr [63:32]
        g0[2] = (unsigned)ga;                         // global_addr [95:64]
        g0[3] = ((unsigned)(ga >> 32) & 0x01FFFFFFu)  // global_addr [120:96]
                | 0x80000000u;                        // type=2 ("image") [127:126]
        u32x8 g1;
        g1[0] = 0x00020000u;        // workgroup_mask=0, data_size=2 (4B)
        g1[1] = (Ln & 0xFFFFu) << 16; // tensor_dim0[15:0] in bits 63:48
        g1[2] = ((Ln >> 16) & 0xFFFFu) | (1u << 16);  // tensor_dim0[31:16], tensor_dim1=1
        g1[3] = (Ln & 0xFFFFu) << 16; // tensor_dim1[31:16]=0, tile_dim0=4096
        g1[4] = 0u;                  // tile_dim1=0 (unused), tile_dim2=0
        g1[5] = (unsigned)Ln;        // tensor_dim0_stride[31:0]
        g1[6] = 0u;                  // stride[47:32]=0, tensor_dim1_stride[15:0]=0
        g1[7] = 0u;
        asm volatile("tensor_load_to_lds %0, %1" :: "s"(g0), "s"(g1) : "memory");
        __builtin_amdgcn_s_wait_tensorcnt(0);
    }
    __syncthreads();                 // row now resident in im[], natural order

    // --- mean/var + bit-reversed scatter im -> re ---
    float s = 0.f, s2 = 0.f;
    for (int l = tid; l < Ln; l += 512) {
        float v = im[l];
        s += v; s2 += v * v;
        re[(int)(__brev((unsigned)l) >> 20)] = v;    // 12-bit reverse
    }
    r1[tid] = s; r2[tid] = s2;
    __syncthreads();                 // re[] complete; im[] reads done
    for (int l = tid; l < Ln; l += 512) im[l] = 0.f;
    for (int st = 256; st > 0; st >>= 1) {
        __syncthreads();
        if (tid < st) { r1[tid] += r1[tid + st]; r2[tid] += r2[tid + st]; }
    }
    __syncthreads();
    // radix-2 DIT, 12 stages
    for (int stg = 0; stg < 12; ++stg) {
        const int half = 1 << stg;
        const float wfac = -6.283185307179586f / (float)(half << 1);
        for (int j = tid; j < (Ln / 2); j += 512) {
            int pos = j & (half - 1);
            int i0  = ((j >> stg) << (stg + 1)) + pos;
            int i1  = i0 + half;
            float sw, cw;
            __sincosf(wfac * (float)pos, &sw, &cw);
            float xr = re[i1], xi = im[i1];
            float tr = cw * xr - sw * xi;
            float ti = cw * xi + sw * xr;
            re[i1] = re[i0] - tr; im[i1] = im[i0] - ti;
            re[i0] += tr;         im[i0] += ti;
        }
        __syncthreads();
    }
    for (int f = tid; f < Fn; f += 512) {
        fre[(size_t)row * Fn + f] = re[f];
        fim[(size_t)row * Fn + f] = im[f];
    }
    if (tid == 0) {
        float mean = r1[0] * (1.0f / Ln);
        float var  = r2[0] * (1.0f / Ln) - mean * mean;
        meanvar[row * 2 + 0] = mean;
        meanvar[row * 2 + 1] = var;
    }
}

// ---------------------------------------------------------------------------
// 3) mag[b,f] = mean_n |freq[b,n,f]|   (prefetch the strided walk)
// ---------------------------------------------------------------------------
__global__ __launch_bounds__(256) void mag_kernel(const float* __restrict__ fre,
                                                  const float* __restrict__ fim,
                                                  float* __restrict__ mag) {
    const int b = blockIdx.y;
    const int f = blockIdx.x * 256 + threadIdx.x;
    if (f >= Fn) return;
    const float* pr = fre + (size_t)b * Nn * Fn + f;
    const float* pi = fim + (size_t)b * Nn * Fn + f;
    float s = 0.f;
    for (int n = 0; n < Nn; ++n) {
        if (n + 4 < Nn) {
            __builtin_prefetch(pr + (size_t)(n + 4) * Fn, 0, 0);
            __builtin_prefetch(pi + (size_t)(n + 4) * Fn, 0, 0);
        }
        float r = pr[(size_t)n * Fn], q = pi[(size_t)n * Fn];
        s += sqrtf(r * r + q * q);
    }
    mag[b * Fn + f] = s * (1.0f / Nn);
}

// ---------------------------------------------------------------------------
// 4) Band boundaries -> integer bin cut points idx[0..8]
// ---------------------------------------------------------------------------
__global__ void band_idx_kernel(const float* __restrict__ bb, int* __restrict__ idx) {
    if (threadIdx.x != 0 || blockIdx.x != 0) return;
    float v[En - 1];
    for (int i = 0; i < En - 1; ++i) v[i] = 1.0f / (1.0f + __expf(-bb[i]));
    for (int i = 1; i < En - 1; ++i) {          // insertion sort (7 elems)
        float key = v[i]; int j = i - 1;
        while (j >= 0 && v[j] > key) { v[j + 1] = v[j]; --j; }
        v[j + 1] = key;
    }
    idx[0] = 0;
    for (int i = 0; i < En - 1; ++i) {
        int t = (int)(v[i] * (float)Fn);        // truncation, as .astype(int32)
        t = t < 0 ? 0 : (t > Fn ? Fn : t);
        idx[i + 1] = t;
    }
    idx[En] = Fn;
}

// ---------------------------------------------------------------------------
// 5a) mag f32 -> f16, K zero-padded to KP
// ---------------------------------------------------------------------------
__global__ __launch_bounds__(256) void mag_to_f16(const float* __restrict__ mag,
                                                  _Float16* __restrict__ magH) {
    int i = blockIdx.x * 256 + threadIdx.x;       // over Bn*KP
    if (i >= Bn * KP) return;
    int m = i / KP, k = i - m * KP;
    magH[i] = (k < Fn) ? (_Float16)mag[m * Fn + k] : (_Float16)0.0f;
}

// ---------------------------------------------------------------------------
// 5b) W1[k][n] f32 -> w1T[n][k] f16, zero-padded to [NP][KP]
// ---------------------------------------------------------------------------
__global__ __launch_bounds__(256) void w1_transpose_f16(const float* __restrict__ w1,
                                                        _Float16* __restrict__ w1T) {
    __shared__ float tile[32][33];
    const int k0 = blockIdx.x * 32;               // 65 tiles -> covers 2080
    const int n0 = blockIdx.y * 32;               // 65 tiles -> covers 2080 (>NP)
    const int tx = threadIdx.x, ty = threadIdx.y; // 32 x 8
    #pragma unroll
    for (int i = 0; i < 32; i += 8) {
        int k = k0 + ty + i, n = n0 + tx;
        tile[ty + i][tx] = (k < Fn && n < Fn) ? w1[(size_t)k * Fn + n] : 0.0f;
    }
    __syncthreads();
    #pragma unroll
    for (int i = 0; i < 32; i += 8) {
        int n = n0 + ty + i, k = k0 + tx;         // k < 2080 always
        if (n < NP) w1T[(size_t)n * KP + k] = (_Float16)tile[tx][ty + i];
    }
}

// ---------------------------------------------------------------------------
// 5c) h = ReLU(mag @ W1 + b1) via V_WMMA_F32_16X16X32_F16.
//     One wave per 16x16 D tile; unconditional b128 fragment loads.
// ---------------------------------------------------------------------------
union FragH { v16h h; u32x4 q[2]; };

__global__ __launch_bounds__(32) void gate_gemm1_wmma(const _Float16* __restrict__ magH,
                                                      const _Float16* __restrict__ w1T,
                                                      const float* __restrict__ b1,
                                                      float* __restrict__ h) {
    const int nt   = blockIdx.x;          // 0..128
    const int mt   = blockIdx.y;          // 0..3
    const int lane = threadIdx.x;         // wave32
    const int mrow = mt * 16 + (lane & 15);
    const int ncol = nt * 16 + (lane & 15);
    const int hi   = lane >> 4;           // half-wave select
    const _Float16* pa = magH + (size_t)mrow * KP + hi * 8;    // + kk
    const _Float16* pb = w1T  + (size_t)ncol * KP + hi * 8;    // + kk
    v8f acc = {};
    for (int kk = 0; kk < KP; kk += 32) {
        FragH fa, fb;
        fa.q[0] = *(const u32x4*)(pa + kk);        // K = kk+kbaseA+0..7
        fa.q[1] = *(const u32x4*)(pa + kk + 16);   // K = kk+kbaseA+16..23
        fb.q[0] = *(const u32x4*)(pb + kk);        // K = kk+kbaseB+0..7
        fb.q[1] = *(const u32x4*)(pb + kk + 16);   // K = kk+kbaseB+16..23
        acc = __builtin_amdgcn_wmma_f32_16x16x32_f16(
            /*neg_a=*/false, fa.h, /*neg_b=*/false, fb.h,
            /*c_mod=*/(short)0, acc, /*reuse_a=*/false, /*reuse_b=*/false);
    }
    if (ncol < Fn) {
        float bias = b1[ncol];
        #pragma unroll
        for (int r = 0; r < 8; ++r) {
            int m = mt * 16 + hi * 8 + r;          // D: M = r + 8*hi, N = lane&15
            float v = acc[r] + bias;
            h[(size_t)m * Fn + ncol] = v > 0.f ? v : 0.f;
        }
    }
}

// ---------------------------------------------------------------------------
// 6) logits = h @ W2 + b2 ; weights = softmax(logits)  (one block per b)
// ---------------------------------------------------------------------------
__global__ __launch_bounds__(256) void gate2_softmax(const float* __restrict__ h,
                                                     const float* __restrict__ w2,
                                                     const float* __restrict__ b2,
                                                     float* __restrict__ wts) {
    const int b = blockIdx.x;
    const int tid = threadIdx.x;
    const int e = tid >> 5, lane = tid & 31;     // one wave per expert
    float p = 0.f;
    for (int k = lane; k < Fn; k += 32) p += h[(size_t)b * Fn + k] * w2[k * En + e];
    __shared__ float red[256];
    __shared__ float logits[En];
    red[tid] = p;
    __syncthreads();
    for (int s = 16; s > 0; s >>= 1) {
        if (lane < s) red[tid] += red[tid + s];
        __syncthreads();
    }
    if (lane == 0) logits[e] = red[tid] + b2[e];
    __syncthreads();
    if (tid == 0) {
        float mx = logits[0];
        for (int i = 1; i < En; ++i) mx = fmaxf(mx, logits[i]);
        float ex[En], s = 0.f;
        for (int i = 0; i < En; ++i) { ex[i] = __expf(logits[i] - mx); s += ex[i]; }
        float inv = 1.0f / s;
        for (int i = 0; i < En; ++i) wts[b * En + i] = ex[i] * inv;
    }
}

// ---------------------------------------------------------------------------
// 7) Apply weighted band mask, inverse FFT (Hermitian reconstruction),
//    rescale by sqrt(var+eps) and add mean. Writes recN[B,N,L].
// ---------------------------------------------------------------------------
__global__ __launch_bounds__(512) void ifft_apply(const float* __restrict__ fre,
                                                  const float* __restrict__ fim,
                                                  const float* __restrict__ wts,
                                                  const int*   __restrict__ idx,
                                                  const float* __restrict__ meanvar,
                                                  float* __restrict__ recN) {
    __shared__ float re[Ln], im[Ln];
    __shared__ float sw8[En];
    __shared__ int   sidx[En + 1];
    const int row = blockIdx.x;
    const int b   = row >> 7;           // row / N
    const int tid = threadIdx.x;
    if (tid < En)     sw8[tid]  = wts[b * En + tid];
    if (tid < En + 1) sidx[tid] = idx[tid];
    __syncthreads();
    const float* pr = fre + (size_t)row * Fn;
    const float* pi = fim + (size_t)row * Fn;
    for (int l = tid; l < Ln; l += 512) {
        int f = (l <= Ln / 2) ? l : Ln - l;
        float wm = 0.f;
        #pragma unroll
        for (int e = 0; e < En; ++e)
            wm += (f >= sidx[e] && f < sidx[e + 1]) ? sw8[e] : 0.f;
        float vr = pr[f] * wm;
        float vi = pi[f] * wm;
        if (l > Ln / 2) vi = -vi;       // conjugate symmetry
        int rb = (int)(__brev((unsigned)l) >> 20);
        re[rb] = vr; im[rb] = vi;
    }
    __syncthreads();
    for (int stg = 0; stg < 12; ++stg) {
        const int half = 1 << stg;
        const float wfac = 6.283185307179586f / (float)(half << 1);   // inverse: +angle
        for (int j = tid; j < (Ln / 2); j += 512) {
            int pos = j & (half - 1);
            int i0  = ((j >> stg) << (stg + 1)) + pos;
            int i1  = i0 + half;
            float sw, cw;
            __sincosf(wfac * (float)pos, &sw, &cw);
            float xr = re[i1], xi = im[i1];
            float tr = cw * xr - sw * xi;
            float ti = cw * xi + sw * xr;
            re[i1] = re[i0] - tr; im[i1] = im[i0] - ti;
            re[i0] += tr;         im[i0] += ti;
        }
        __syncthreads();
    }
    const float mean = meanvar[row * 2 + 0];
    const float var  = meanvar[row * 2 + 1];
    const float sc   = sqrtf(var + 1e-6f) * (1.0f / Ln);
    float* dst = recN + (size_t)row * Ln;
    for (int l = tid; l < Ln; l += 512)
        dst[l] = re[l] * sc + mean;
}

// ---------------------------------------------------------------------------
// 8) out[b,l,n] = x[b,l,n] + recN[b,n,l]   (residual + transpose back)
// ---------------------------------------------------------------------------
__global__ __launch_bounds__(256) void final_add_transpose(const float* __restrict__ recN,
                                                           const float* __restrict__ x,
                                                           float* __restrict__ out) {
    __shared__ float tile[32][33];
    const int b  = blockIdx.z;
    const int l0 = blockIdx.x * 32;
    const int n0 = blockIdx.y * 32;
    const int tx = threadIdx.x, ty = threadIdx.y;
    #pragma unroll
    for (int i = 0; i < 32; i += 8) {
        int n = n0 + ty + i, l = l0 + tx;
        tile[ty + i][tx] = recN[((size_t)b * Nn + n) * Ln + l];   // tile[n-n0][l-l0]
    }
    __syncthreads();
    #pragma unroll
    for (int i = 0; i < 32; i += 8) {
        int l = l0 + ty + i, n = n0 + tx;
        size_t o = ((size_t)b * Ln + l) * Nn + n;
        out[o] = x[o] + tile[tx][ty + i];
    }
}

// ---------------------------------------------------------------------------
// Workspace layout (bytes, OFF_H and later offsets are not 16B aligned)
// ---------------------------------------------------------------------------
static constexpr size_t SZ_XN   = (size_t)Bn * Nn * Ln * 4;   // 134,217,728 (reused as recN)
static constexpr size_t SZ_FRQ  = (size_t)Bn * Nn * Fn * 4;   //  67,141,632
static constexpr size_t OFF_XN   = 0;
static constexpr size_t OFF_FRE  = OFF_XN  + SZ_XN;
static constexpr size_t OFF_FIM  = OFF_FRE + SZ_FRQ;
static constexpr size_t OFF_MV   = OFF_FIM + SZ_FRQ;          // ROWS*2 floats
static constexpr size_t OFF_MAG  = OFF_MV  + (size_t)ROWS * 2 * 4;
static constexpr size_t OFF_H    = OFF_MAG + (size_t)Bn * Fn * 4 + 12;   // 12 B pad; OFF_H is 12 mod 16
static constexpr size_t OFF_W    = OFF_H   + (size_t)Bn * Fn * 4 + 12;   // 12 B pad; OFF_W is 8 mod 16
static constexpr size_t OFF_IDX  = OFF_W   + (size_t)Bn * En * 4;
static constexpr size_t OFF_MAGH = OFF_IDX + 64;                         // f16 [Bn][KP]
static constexpr size_t OFF_W1T  = OFF_MAGH + (size_t)Bn * KP * 2;       // f16 [NP][KP]

extern "C" void kernel_launch(void* const* d_in, const int* in_sizes, int n_in,
                              void* d_out, int out_size, void* d_ws, size_t ws_size,
                              hipStream_t stream) {
    const float* x  = (const float*)d_in[0];
    const float* bb = (const float*)d_in[1];
    const float* w1 = (const float*)d_in[2];
    const float* b1 = (const float*)d_in[3];
    const float* w2 = (const float*)d_in[4];
    const float* b2 = (const float*)d_in[5];
    float* out = (float*)d_out;

    char* ws = (char*)d_ws;
    float*     xN   = (float*)(ws + OFF_XN);    // reused as recN after fwd FFT
    float*     fre  = (float*)(ws + OFF_FRE);
    float*     fim  = (float*)(ws + OFF_FIM);
    float*     mv   = (float*)(ws + OFF_MV);
    float*     mag  = (float*)(ws + OFF_MAG);
    float*     h    = (float*)(ws + OFF_H);
    float*     wts  = (float*)(ws + OFF_W);
    int*       idx  = (int*)  (ws + OFF_IDX);
    _Float16*  magH = (_Float16*)(ws + OFF_MAGH);
    _Float16*  w1T  = (_Float16*)(ws + OFF_W1T);

    transpose_xn       <<<dim3(Ln / 32, Nn / 32, Bn), dim3(32, 8), 0, stream>>>(x, xN);
    fft_fwd            <<<ROWS, 512, 0, stream>>>(xN, fre, fim, mv);
    mag_kernel         <<<dim3((Fn + 255) / 256, Bn), 256, 0, stream>>>(fre, fim, mag);
    band_idx_kernel    <<<1, 32, 0, stream>>>(bb, idx);
    mag_to_f16         <<<(Bn * KP + 255) / 256, 256, 0, stream>>>(mag, magH);
    w1_transpose_f16   <<<dim3(KP / 32, KP / 32), dim3(32, 8), 0, stream>>>(w1, w1T);
    gate_gemm1_wmma    <<<dim3(NT, Bn / 16), 32, 0, stream>>>(magH, w1T, b1, h);
    gate2_softmax      <<<Bn, 256, 0, stream>>>(h, w2, b2, wts);
    ifft_apply         <<<ROWS, 512, 0, stream>>>(fre, fim, wts, idx, mv, xN /* recN */);
    final_add_transpose<<<dim3(Ln / 32, Nn / 32, Bn), dim3(32, 8), 0, stream>>>(xN, x, out);
}
